// AVIMBlock_74612171866161
// MI455X (gfx1250) — compile-verified, matrix operand order checked by reading, not run on hardware
//
#include <hip/hip_runtime.h>
#include <hip/hip_bf16.h>

typedef __bf16 bf16_t;
typedef __attribute__((ext_vector_type(16))) __bf16 v16bf;
typedef __attribute__((ext_vector_type(8)))  float  v8f;
typedef __attribute__((ext_vector_type(4)))  int    v4i;

#define DMODEL 768
#define ROWS   4096   // B*N = 2*2048
#define MLPH   3072

// ---------------------------------------------------------------------------
// Async global->LDS 16B copy (CDNA5 GLOBAL_LOAD_ASYNC_TO_LDS_B128, ASYNCcnt),
// with a synchronous fallback if the builtin is absent on this toolchain.
// Builtin wants v4i* in AS1 (global) / AS3 (LDS) per the round-2 diagnostic.
// ---------------------------------------------------------------------------
static __device__ __forceinline__ void cp_g2l_16(const bf16_t* g, bf16_t* l) {
#if __has_builtin(__builtin_amdgcn_global_load_async_to_lds_b128)
  __builtin_amdgcn_global_load_async_to_lds_b128(
      (__attribute__((address_space(1))) v4i*)(v4i*)g,
      (__attribute__((address_space(3))) v4i*)(v4i*)l, 0, 0);
#else
  *(uint4*)l = *(const uint4*)g;
#endif
}
static __device__ __forceinline__ void cp_g2l_wait() {
#if __has_builtin(__builtin_amdgcn_global_load_async_to_lds_b128)
#if __has_builtin(__builtin_amdgcn_s_wait_asynccnt)
  __builtin_amdgcn_s_wait_asynccnt(0);
#else
  asm volatile("s_wait_asynccnt 0" ::: "memory");
#endif
#endif
}

// ---------------------------------------------------------------------------
// Fragment loader: per ISA 7.12.2 (16-bit A 16x32 / B stored N-major in LDS):
// lane l -> row/col (l&15), kb = (l>=16)?8:0, VGPR j: K = kb+2j (j<4) else
// 16+kb+2(j-4). Pairs are contiguous -> compiler merges into ds_load_b128.
// ---------------------------------------------------------------------------
static __device__ __forceinline__ v16bf ld_frag(const bf16_t* p, int row,
                                                int stride, int kb, int kadd) {
  union { unsigned u[8]; v16bf v; } f;
#pragma unroll
  for (int j = 0; j < 8; ++j) {
    int ko = kadd + ((j < 4) ? (kb + 2 * j) : (16 + kb + 2 * (j - 4)));
    f.u[j] = *(const unsigned*)(p + row * stride + ko);
  }
  return f.v;
}

// ---------------------------------------------------------------------------
// Row LayerNorm (768 cols) + cast to bf16.  One block per row.
// ---------------------------------------------------------------------------
__global__ __launch_bounds__(256) void ln_cast_kernel(
    const float* __restrict__ x, const float* __restrict__ g,
    const float* __restrict__ b, bf16_t* __restrict__ out) {
  __shared__ float red[256];
  int row = blockIdx.x, tid = threadIdx.x;
  const float* xr = x + (size_t)row * DMODEL;
  float v0 = xr[tid], v1 = xr[tid + 256], v2 = xr[tid + 512];
  red[tid] = v0 + v1 + v2;
  __syncthreads();
  for (int s = 128; s > 0; s >>= 1) {
    if (tid < s) red[tid] += red[tid + s];
    __syncthreads();
  }
  float mu = red[0] * (1.0f / DMODEL);
  __syncthreads();
  float d0 = v0 - mu, d1 = v1 - mu, d2 = v2 - mu;
  red[tid] = d0 * d0 + d1 * d1 + d2 * d2;
  __syncthreads();
  for (int s = 128; s > 0; s >>= 1) {
    if (tid < s) red[tid] += red[tid + s];
    __syncthreads();
  }
  float rstd = rsqrtf(red[0] * (1.0f / DMODEL) + 1e-5f);
  bf16_t* o = out + (size_t)row * DMODEL;
  o[tid]       = (bf16_t)(d0 * rstd * g[tid]       + b[tid]);
  o[tid + 256] = (bf16_t)(d1 * rstd * g[tid + 256] + b[tid + 256]);
  o[tid + 512] = (bf16_t)(d2 * rstd * g[tid + 512] + b[tid + 512]);
}

// Plain f32 -> bf16 cast
__global__ void cast_kernel(const float* __restrict__ x, bf16_t* __restrict__ o,
                            int n) {
  int i = blockIdx.x * blockDim.x + threadIdx.x;
  int stride = gridDim.x * blockDim.x;
  for (; i < n; i += stride) o[i] = (bf16_t)x[i];
}

// W[K][N] f32 -> Wt[N][K] bf16 (tiled transpose through LDS)
__global__ __launch_bounds__(256) void tcast_kernel(
    const float* __restrict__ W, bf16_t* __restrict__ Wt, int K, int N) {
  __shared__ float tile[32][33];
  int nb = blockIdx.x * 32, kb = blockIdx.y * 32;
  int tx = threadIdx.x & 31, ty = threadIdx.x >> 5;
#pragma unroll
  for (int i = 0; i < 32; i += 8)
    tile[ty + i][tx] = W[(size_t)(kb + ty + i) * N + nb + tx];
  __syncthreads();
#pragma unroll
  for (int i = 0; i < 32; i += 8)
    Wt[(size_t)(nb + ty + i) * K + kb + tx] = (bf16_t)tile[tx][ty + i];
}

// ---------------------------------------------------------------------------
// bf16 WMMA GEMM:  C[M,N] = A[M,K] * Wt[N,K]^T (+bias)(gelu)(+residual)
// Block = 256 thr = 8 waves, tile 128x128, wave tile 32x64 (2x4 WMMA accs).
// K-step 64 per staging round -> 16 WMMAs per barrier pair.
// M,N,K all multiples of 128/64 by construction -> no bounds checks.
// ---------------------------------------------------------------------------
__global__ __launch_bounds__(256) void gemm_bf16_kernel(
    const bf16_t* __restrict__ A, const bf16_t* __restrict__ Wt,
    const float* __restrict__ bias, const float* __restrict__ residual,
    float* __restrict__ outF, bf16_t* __restrict__ outB,
    int M, int N, int K, int dogelu) {
  __shared__ bf16_t As[128 * 72];   // stride 72: 16B-aligned, conflict-free
  __shared__ bf16_t Bs[128 * 72];
  int tid = threadIdx.x;
  int w = tid >> 5, lane = tid & 31;
  int wr = w >> 1, wc = w & 1;
  int half = lane >> 4, lr = lane & 15;
  int kb = half * 8;
  int rowBase = blockIdx.x * 128, colBase = blockIdx.y * 128;

  // staging assignment: thread -> rows {r0, r0+32, r0+64, r0+96}, 8-elem col c8
  int r0 = tid >> 3;
  int c8 = (tid & 7) * 8;
  const bf16_t* gA = A + (size_t)(rowBase + r0) * K + c8;
  const bf16_t* gB = Wt + (size_t)(colBase + r0) * K + c8;
  bf16_t* lA = &As[r0 * 72 + c8];
  bf16_t* lB = &Bs[r0 * 72 + c8];
  const size_t gstep = (size_t)32 * K;
  const int lstep = 32 * 72;

  v8f zero = {0.f, 0.f, 0.f, 0.f, 0.f, 0.f, 0.f, 0.f};
  v8f acc[2][4];
#pragma unroll
  for (int mt = 0; mt < 2; ++mt)
#pragma unroll
    for (int nt = 0; nt < 4; ++nt) acc[mt][nt] = zero;

  for (int k0 = 0; k0 < K; k0 += 64) {
    __syncthreads();
#pragma unroll
    for (int q = 0; q < 4; ++q) {
      cp_g2l_16(gA + q * gstep, lA + q * lstep);
      cp_g2l_16(gB + q * gstep, lB + q * lstep);
    }
    cp_g2l_wait();
    __syncthreads();
    gA += 64;
    gB += 64;
#pragma unroll
    for (int ks = 0; ks < 2; ++ks) {
      v16bf af[2], bfr[4];
#pragma unroll
      for (int mt = 0; mt < 2; ++mt)
        af[mt] = ld_frag(As, wr * 32 + mt * 16 + lr, 72, kb, ks * 32);
#pragma unroll
      for (int nt = 0; nt < 4; ++nt)
        bfr[nt] = ld_frag(Bs, wc * 64 + nt * 16 + lr, 72, kb, ks * 32);
#pragma unroll
      for (int mt = 0; mt < 2; ++mt)
#pragma unroll
        for (int nt = 0; nt < 4; ++nt)
          acc[mt][nt] = __builtin_amdgcn_wmma_f32_16x16x32_bf16(
              false, af[mt], false, bfr[nt], (short)0, acc[mt][nt], false, false);
    }
  }

  // epilogue — C-frag layout: VGPR i -> M = i + 8*half, N = lr
#pragma unroll
  for (int mt = 0; mt < 2; ++mt) {
#pragma unroll
    for (int nt = 0; nt < 4; ++nt) {
      int n = colBase + wc * 64 + nt * 16 + lr;
      float bv = bias ? bias[n] : 0.0f;
#pragma unroll
      for (int i = 0; i < 8; ++i) {
        int m = rowBase + wr * 32 + mt * 16 + half * 8 + i;
        float v = acc[mt][nt][i] + bv;
        if (dogelu) v = 0.5f * v * (1.0f + erff(v * 0.70710678118f));
        if (residual) v += residual[(size_t)m * N + n];
        if (outF) outF[(size_t)m * N + n] = v;
        if (outB) outB[(size_t)m * N + n] = (bf16_t)v;
      }
    }
  }
}

// ---------------------------------------------------------------------------
// Flash-style re-softmax cross attention. Grid (qtile=2, head=12, bt=16).
// Block 256 thr = 8 waves, each wave owns 16 q-rows of a 128-row tile.
// logits = -(q.k)/8 ; invalid neighbor segments are skipped entirely
// (equivalent to the -inf masking in the reference).
// ---------------------------------------------------------------------------
__global__ __launch_bounds__(256) void attn_kernel(
    const bf16_t* __restrict__ Q, const bf16_t* __restrict__ Kk,
    const bf16_t* __restrict__ V, bf16_t* __restrict__ O) {
  __shared__ bf16_t Qs[128 * 72];      // 128 q-rows x 64 d
  __shared__ bf16_t Ks[32 * 72];       // 32 keys  x 64 d
  __shared__ bf16_t Vt[64 * 40];       // 64 d x 32 keys (transposed)
  __shared__ bf16_t Ps[8 * 16 * 40];   // per-wave P patch (16 q x 32 keys)

  int bt = blockIdx.z, h = blockIdx.y, qt = blockIdx.x;
  int b = bt >> 3, t = bt & 7;
  int tid = threadIdx.x, w = tid >> 5, lane = tid & 31;
  int half = lane >> 4, lr = lane & 15, kbs = half * 8;
  size_t qRow0 = (size_t)bt * 256 + (size_t)qt * 128;

  // stage Q tile (async): 128 rows x 8 chunks, 4 chunks per thread
#pragma unroll
  for (int q = 0; q < 4; ++q) {
    int r = (tid >> 3) + q * 32, c8 = (tid & 7) * 8;
    cp_g2l_16(&Q[(qRow0 + r) * DMODEL + h * 64 + c8], &Qs[r * 72 + c8]);
  }

  v8f zero = {0.f, 0.f, 0.f, 0.f, 0.f, 0.f, 0.f, 0.f};
  v8f o[4];
#pragma unroll
  for (int nt = 0; nt < 4; ++nt) o[nt] = zero;
  float mrow[8], lsum[8];
#pragma unroll
  for (int i = 0; i < 8; ++i) { mrow[i] = -3.0e38f; lsum[i] = 0.0f; }

  for (int j = t - 1; j <= t + 1; ++j) {
    if (j < 0 || j >= 8) continue;               // skip invalid segment
    size_t kRow0 = (size_t)b * 2048 + (size_t)j * 256;
    for (int kt = 0; kt < 8; ++kt) {             // 32 keys per subtile
      size_t kr = kRow0 + kt * 32;
      __syncthreads();
      { // stage K subtile (row-major: directly the B-frag layout), async
        int r = tid >> 3, c8 = (tid & 7) * 8;
        cp_g2l_16(&Kk[(kr + r) * DMODEL + h * 64 + c8], &Ks[r * 72 + c8]);
      }
      { // stage V subtile transposed: Vt[d][key] (manual, needs shuffle)
        int key = tid >> 3, d8 = (tid & 7) * 8;
        union { uint4 q; bf16_t e[8]; } vv;
        vv.q = *(const uint4*)&V[(kr + key) * DMODEL + h * 64 + d8];
#pragma unroll
        for (int i = 0; i < 8; ++i) Vt[(d8 + i) * 40 + key] = vv.e[i];
      }
      cp_g2l_wait();
      __syncthreads();

      // S = -(Q.K^T)/8   (two 16-key n-tiles, two 32-d k-steps)
      v16bf aq0 = ld_frag(Qs, w * 16 + lr, 72, kbs, 0);
      v16bf aq1 = ld_frag(Qs, w * 16 + lr, 72, kbs, 32);
      float sv[2][8];
#pragma unroll
      for (int nt = 0; nt < 2; ++nt) {
        v8f accs = zero;
        v16bf bk0 = ld_frag(Ks, nt * 16 + lr, 72, kbs, 0);
        v16bf bk1 = ld_frag(Ks, nt * 16 + lr, 72, kbs, 32);
        accs = __builtin_amdgcn_wmma_f32_16x16x32_bf16(false, aq0, false, bk0,
                                                       (short)0, accs, false, false);
        accs = __builtin_amdgcn_wmma_f32_16x16x32_bf16(false, aq1, false, bk1,
                                                       (short)0, accs, false, false);
#pragma unroll
        for (int i = 0; i < 8; ++i) sv[nt][i] = accs[i] * (-0.125f);
      }

      // online softmax (row = i + 8*half lives in one 16-lane group)
#pragma unroll
      for (int i = 0; i < 8; ++i) {
        float v = fmaxf(sv[0][i], sv[1][i]);
#pragma unroll
        for (int mk = 1; mk < 16; mk <<= 1) v = fmaxf(v, __shfl_xor(v, mk, 32));
        float nm = fmaxf(mrow[i], v);
        float alpha = __expf(mrow[i] - nm);
        float p0 = __expf(sv[0][i] - nm);
        float p1 = __expf(sv[1][i] - nm);
        float rs = p0 + p1;
#pragma unroll
        for (int mk = 1; mk < 16; mk <<= 1) rs += __shfl_xor(rs, mk, 32);
        lsum[i] = lsum[i] * alpha + rs;
        mrow[i] = nm;
#pragma unroll
        for (int nt = 0; nt < 4; ++nt) o[nt][i] *= alpha;
        // re-layout C-frag -> A-frag through wave-private LDS patch
        Ps[w * 640 + (i + half * 8) * 40 + lr]      = (bf16_t)p0;
        Ps[w * 640 + (i + half * 8) * 40 + 16 + lr] = (bf16_t)p1;
      }

      // O += P[16,32] x V[32,64]
      v16bf ap = ld_frag(Ps + w * 640, lr, 40, kbs, 0);
#pragma unroll
      for (int nt = 0; nt < 4; ++nt) {
        v16bf bv = ld_frag(Vt, nt * 16 + lr, 40, kbs, 0);
        o[nt] = __builtin_amdgcn_wmma_f32_16x16x32_bf16(false, ap, false, bv,
                                                        (short)0, o[nt], false, false);
      }
    }
  }

  // normalize and store (attO row-major [4096, 768], col = h*64+d)
#pragma unroll
  for (int nt = 0; nt < 4; ++nt)
#pragma unroll
    for (int i = 0; i < 8; ++i) {
      size_t row = qRow0 + w * 16 + half * 8 + i;
      O[row * DMODEL + h * 64 + nt * 16 + lr] = (bf16_t)(o[nt][i] / lsum[i]);
    }
}

// ---------------------------------------------------------------------------
extern "C" void kernel_launch(void* const* d_in, const int* in_sizes, int n_in,
                              void* d_out, int out_size, void* d_ws, size_t ws_size,
                              hipStream_t stream) {
  (void)in_sizes; (void)n_in; (void)out_size; (void)ws_size;
  const float* video = (const float*)d_in[0];
  const float* audio = (const float*)d_in[1];

  char* ws = (char*)d_ws;
  size_t off = 0;
  auto alloc = [&](size_t bytes) -> char* {
    char* p = ws + off;
    off = (off + bytes + 255) & ~(size_t)255;
    return p;
  };
  const size_t SZ_DD  = (size_t)768 * 768 * 2;      // bf16 768x768
  const size_t SZ_DH  = (size_t)768 * 3072 * 2;     // bf16 768x3072
  const size_t SZ_ACT = (size_t)ROWS * DMODEL * 2;  // bf16 4096x768

  bf16_t* wqT = (bf16_t*)alloc(SZ_DD);
  bf16_t* wkT = (bf16_t*)alloc(SZ_DD);
  bf16_t* wvT = (bf16_t*)alloc(SZ_DD);
  bf16_t* woT = (bf16_t*)alloc(SZ_DD);
  bf16_t* w1T = (bf16_t*)alloc(SZ_DH);
  bf16_t* w2T = (bf16_t*)alloc(SZ_DH);
  // contiguous group [qn, kn, Qb, Kb]: all dead by MLP time -> h1 aliases it
  bf16_t* qn  = (bf16_t*)alloc(SZ_ACT);
  bf16_t* kn  = (bf16_t*)alloc(SZ_ACT);
  bf16_t* Qb  = (bf16_t*)alloc(SZ_ACT);
  bf16_t* Kb  = (bf16_t*)alloc(SZ_ACT);
  bf16_t* kr  = (bf16_t*)alloc(SZ_ACT);
  bf16_t* Vb  = (bf16_t*)alloc(SZ_ACT);
  bf16_t* aO  = (bf16_t*)alloc(SZ_ACT);
  float*  xbuf = (float*)alloc((size_t)ROWS * DMODEL * 4);
  bf16_t* xn  = (bf16_t*)alloc(SZ_ACT);
  bf16_t* h1  = qn;  // 4*SZ_ACT == ROWS*MLPH*2 exactly

  for (int blk = 0; blk < 2; ++blk) {
    int pb = 2 + blk * 18;  // param dict flattened in insertion order
    const float* qf = blk ? audio : video;
    const float* kf = blk ? video : audio;
    const float* nq_g = (const float*)d_in[pb + 0];
    const float* nq_b = (const float*)d_in[pb + 1];
    const float* nk_g = (const float*)d_in[pb + 2];
    const float* nk_b = (const float*)d_in[pb + 3];
    const float* n2_g = (const float*)d_in[pb + 4];
    const float* n2_b = (const float*)d_in[pb + 5];
    const float* wq = (const float*)d_in[pb + 6];
    const float* bq = (const float*)d_in[pb + 7];
    const float* wk = (const float*)d_in[pb + 8];
    const float* bk = (const float*)d_in[pb + 9];
    const float* wv = (const float*)d_in[pb + 10];
    const float* bv = (const float*)d_in[pb + 11];
    const float* wo = (const float*)d_in[pb + 12];
    const float* bo = (const float*)d_in[pb + 13];
    const float* w1 = (const float*)d_in[pb + 14];
    const float* b1 = (const float*)d_in[pb + 15];
    const float* w2 = (const float*)d_in[pb + 16];
    const float* b2 = (const float*)d_in[pb + 17];
    float* outp = (float*)d_out + (size_t)blk * ROWS * DMODEL;

    tcast_kernel<<<dim3(24, 24), 256, 0, stream>>>(wq, wqT, 768, 768);
    tcast_kernel<<<dim3(24, 24), 256, 0, stream>>>(wk, wkT, 768, 768);
    tcast_kernel<<<dim3(24, 24), 256, 0, stream>>>(wv, wvT, 768, 768);
    tcast_kernel<<<dim3(24, 24), 256, 0, stream>>>(wo, woT, 768, 768);
    tcast_kernel<<<dim3(96, 24), 256, 0, stream>>>(w1, w1T, 768, 3072);
    tcast_kernel<<<dim3(24, 96), 256, 0, stream>>>(w2, w2T, 3072, 768);

    ln_cast_kernel<<<ROWS, 256, 0, stream>>>(qf, nq_g, nq_b, qn);
    ln_cast_kernel<<<ROWS, 256, 0, stream>>>(kf, nk_g, nk_b, kn);
    cast_kernel<<<1024, 256, 0, stream>>>(kf, kr, ROWS * DMODEL);

    gemm_bf16_kernel<<<dim3(32, 6), 256, 0, stream>>>(
        qn, wqT, bq, nullptr, nullptr, Qb, ROWS, 768, 768, 0);
    gemm_bf16_kernel<<<dim3(32, 6), 256, 0, stream>>>(
        kn, wkT, bk, nullptr, nullptr, Kb, ROWS, 768, 768, 0);
    gemm_bf16_kernel<<<dim3(32, 6), 256, 0, stream>>>(
        kr, wvT, bv, nullptr, nullptr, Vb, ROWS, 768, 768, 0);

    attn_kernel<<<dim3(2, 12, 16), 256, 0, stream>>>(Qb, Kb, Vb, aO);

    // x = qf + attO @ wo + bo   (f32)
    gemm_bf16_kernel<<<dim3(32, 6), 256, 0, stream>>>(
        aO, woT, bo, qf, xbuf, nullptr, ROWS, 768, 768, 0);
    ln_cast_kernel<<<ROWS, 256, 0, stream>>>(xbuf, n2_g, n2_b, xn);
    // h1 = gelu(xn @ w1 + b1)   (bf16, aliases dead qn..Kb region)
    gemm_bf16_kernel<<<dim3(32, 24), 256, 0, stream>>>(
        xn, w1T, b1, nullptr, nullptr, h1, ROWS, 3072, 768, 1);
    // out = x + h1 @ w2 + b2    (f32, straight into d_out slice)
    gemm_bf16_kernel<<<dim3(32, 6), 256, 0, stream>>>(
        h1, w2T, b2, xbuf, outp, nullptr, ROWS, 768, 3072, 0);
  }
}
